// Attention_Bank_11098195493450
// MI455X (gfx1250) — matrix operand order repaired and checked by reading, not compiled, not run on hardware
//
#include <hip/hip_runtime.h>
#include <hip/hip_bf16.h>

// ---------------------------------------------------------------------------
// Problem constants (from reference)
// ---------------------------------------------------------------------------
#define BB    8
#define CC    64
#define HH    128
#define WW    128
#define HWSZ  (HH * WW)        // 16384
#define EE    3
#define NHEAD 16
#define HID   16
#define NQKV  (3 * CC)         // 192

// d_out layout: image [B,C,H,W] | hidden_new [B,16] | logit5 [B,3]
#define OUT_IMG_ELEMS   (BB * CC * HWSZ)      // 8388608
#define OUT_HID_OFF     OUT_IMG_ELEMS
#define OUT_LOGIT_OFF   (OUT_IMG_ELEMS + BB * HID)

// Workspace layout (bytes)
#define OFF_XG     0u                                   // B*C floats      = 2048
#define OFF_LOGIT  2048u                                // B*E floats      = 96
#define OFF_GRAM   4096u                                // 128*24 floats   = 12288
#define OFF_ATTN   16384u                               // 128*16 floats   = 8192
#define OFF_QKV    24576u                               // f16 B*192*HW    = 50331648
#define OFF_U      (24576u + 50331648u)                 // f16 B*64*HW     = 16777216
// total ~67.1 MB (fits comfortably in MI455X's 192 MB L2)

typedef __attribute__((ext_vector_type(16))) _Float16 v16h;
typedef __attribute__((ext_vector_type(8)))  _Float16 v8h;
typedef __attribute__((ext_vector_type(4)))  _Float16 v4h;
typedef __attribute__((ext_vector_type(2)))  _Float16 v2h;
typedef __attribute__((ext_vector_type(8)))  float    v8f;

// LDS activation tiles are pixel-major with padded rows: 64 halves + 8 pad
// -> 144-byte rows (16B-aligned), so every WMMA fragment is a ds_load_b128 run.
#define XROW 72

// ---------------------------------------------------------------------------
// 1) Spatial mean per (b,c): xg[b*C+c] = mean(x[b,c,:,:])
// ---------------------------------------------------------------------------
__global__ __launch_bounds__(256) void mean_kernel(const float* __restrict__ x,
                                                   float* __restrict__ xg) {
  const int bc = blockIdx.x;                       // 0..B*C-1
  const float* p = x + (size_t)bc * HWSZ;
  float s = 0.f;
  for (int i = threadIdx.x; i < HWSZ; i += 256) s += p[i];
  #pragma unroll
  for (int m = 16; m > 0; m >>= 1) s += __shfl_xor(s, m, 32);
  __shared__ float acc;
  if (threadIdx.x == 0) acc = 0.f;
  __syncthreads();
  if ((threadIdx.x & 31) == 0) atomicAdd(&acc, s);
  __syncthreads();
  if (threadIdx.x == 0) xg[bc] = acc * (1.0f / (float)HWSZ);
}

// ---------------------------------------------------------------------------
// 2) Routing MLP: h1 = gelu(r1_w @ [xg; hidden] + r1_b); logit = relu(r3_w@h1+r3_b)
// ---------------------------------------------------------------------------
__global__ void routing_kernel(const float* __restrict__ xg,
                               const float* __restrict__ hidden,
                               const float* __restrict__ r1w,
                               const float* __restrict__ r1b,
                               const float* __restrict__ r3w,
                               const float* __restrict__ r3b,
                               float* __restrict__ out_hidden,
                               float* __restrict__ out_logit,
                               float* __restrict__ logit_ws) {
  __shared__ float hn[BB][HID];
  const int t = threadIdx.x;
  if (t < BB * HID) {
    const int b = t >> 4, o = t & 15;
    const float* wr = r1w + o * (CC + HID);
    float s = r1b[o];
    for (int c = 0; c < CC; ++c)  s += wr[c] * xg[b * CC + c];
    for (int c = 0; c < HID; ++c) s += wr[CC + c] * hidden[b * HID + c];
    // exact GELU (approximate=False)
    const float g = 0.5f * s * (1.0f + erff(s * 0.70710678118654752440f));
    hn[b][o] = g;
    out_hidden[b * HID + o] = g;
  }
  __syncthreads();
  if (t < BB * EE) {
    const int b = t / EE, e = t % EE;
    float s = r3b[e];
    for (int h = 0; h < HID; ++h) s += r3w[e * HID + h] * hn[b][h];
    s = fmaxf(s, 0.f);
    out_logit[b * EE + e] = s;
    logit_ws[b * EE + e]  = s;
  }
}

// ---------------------------------------------------------------------------
// WMMA fragment helpers (contiguous k order k = 16h + s, same for A and B).
//   A 16x32 (MxK): lane row = lane&15; K run:  [(lane<16?0:16) .. +15]
//   B 32x16 (KxN): lane col = lane&15; K run:  [(lane<16?0:16) .. +15]
// With pixel-major LDS rows both become contiguous ds_load_b128 runs.
// ---------------------------------------------------------------------------
__device__ __forceinline__ v16h load_a_frag(const _Float16* wrow, int c0, int hi) {
  const v8h a0 = *(const v8h*)(wrow + c0 + hi);
  const v8h a1 = *(const v8h*)(wrow + c0 + hi + 8);
  return __builtin_shufflevector(a0, a1, 0,1,2,3,4,5,6,7,8,9,10,11,12,13,14,15);
}
__device__ __forceinline__ v16h load_b_frag(const _Float16* xrow, int c0, int kb) {
  const v8h b0 = *(const v8h*)(xrow + c0 + kb);
  const v8h b1 = *(const v8h*)(xrow + c0 + kb + 8);
  return __builtin_shufflevector(b0, b1, 0,1,2,3,4,5,6,7,8,9,10,11,12,13,14,15);
}

// ---------------------------------------------------------------------------
// 3) QKV 1x1-conv GEMM via WMMA f16 -> f32:
//    qkv[b,o,n] = sum_c w[o,c] * x[b,c,n],  o in [0,192), K=64, f16 output.
//    Grid: x = B*HW/128 (128 pixels/block, 8 waves x 16-pixel N-tiles),
//          y = 12 (16-channel M-tiles). Block = 256.
// ---------------------------------------------------------------------------
__global__ __launch_bounds__(256) void qkv_gemm_kernel(const float* __restrict__ x,
                                                       const float* __restrict__ w,
                                                       _Float16* __restrict__ qkv) {
  __shared__ __align__(16) _Float16 Xs[128][XROW];  // pixel-major x tile (f16)
  __shared__ __align__(16) _Float16 Ws[16][CC];     // weight tile (f16)
  const int b    = blockIdx.x >> 7;
  const int pix0 = (blockIdx.x & 127) << 7;
  const int o0   = blockIdx.y << 4;
  const int tid  = threadIdx.x;

  for (int i = tid; i < 16 * CC; i += 256)
    Ws[i >> 6][i & 63] = (_Float16)w[(o0 + (i >> 6)) * CC + (i & 63)];

  // Transposing fill: thread handles (c-group of 4, pixel n). The 4 global
  // loads are n-coalesced across lanes; the LDS store is one ds_store_b64.
  const float* xb = x + (size_t)b * CC * HWSZ + pix0;
  for (int i = tid; i < 16 * 128; i += 256) {
    const int n = i & 127, c4 = (i >> 7) << 2;
    v4h v;
    #pragma unroll
    for (int j = 0; j < 4; ++j)
      v[j] = (_Float16)xb[(size_t)(c4 + j) * HWSZ + n];
    *(v4h*)&Xs[n][c4] = v;
  }
  __syncthreads();

  const int lane = tid & 31;
  const int nloc = (tid >> 5) << 4;     // this wave's 16-pixel subtile
  const int col  = lane & 15;           // A row / B column / D column
  const int hi   = (lane < 16) ? 0 : 8;
  const int kb   = (lane < 16) ? 0 : 16;

  v8f acc = {};
  #pragma unroll
  for (int ks = 0; ks < 2; ++ks) {
    const int c0 = ks * 32;
    const v16h a  = load_a_frag(&Ws[col][0], c0, kb);
    const v16h bf = load_b_frag(&Xs[nloc + col][0], c0, kb);
    acc = __builtin_amdgcn_wmma_f32_16x16x32_f16(false, a, false, bf,
                                                 (short)0, acc, false, false);
  }

  _Float16* q = qkv + (size_t)b * NQKV * HWSZ + (pix0 + nloc + col);
  #pragma unroll
  for (int r = 0; r < 8; ++r)
    q[(size_t)(o0 + r + hi) * HWSZ] = (_Float16)acc[r];
}

// ---------------------------------------------------------------------------
// 4) Gram + norms per (b,head): G[c][d]=sum_n q_c k_d, qq, kk (over n=16384)
// ---------------------------------------------------------------------------
__global__ __launch_bounds__(256) void gram_kernel(const _Float16* __restrict__ qkv,
                                                   float* __restrict__ gram) {
  const int bh = blockIdx.x, b = bh >> 4, head = bh & 15;
  const _Float16* qp = qkv + ((size_t)b * NQKV + head * 4) * HWSZ;
  const _Float16* kp = qp + (size_t)CC * HWSZ;

  float qq[4] = {}, kk[4] = {}, g[4][4] = {};
  for (int n = threadIdx.x * 2; n < HWSZ; n += 512) {
    float qv[4][2], kv[4][2];
    #pragma unroll
    for (int d = 0; d < 4; ++d) {
      const v2h q2 = *(const v2h*)(qp + (size_t)d * HWSZ + n);
      const v2h k2 = *(const v2h*)(kp + (size_t)d * HWSZ + n);
      qv[d][0] = (float)q2[0]; qv[d][1] = (float)q2[1];
      kv[d][0] = (float)k2[0]; kv[d][1] = (float)k2[1];
    }
    #pragma unroll
    for (int p = 0; p < 2; ++p)
      #pragma unroll
      for (int d = 0; d < 4; ++d) {
        qq[d] += qv[d][p] * qv[d][p];
        kk[d] += kv[d][p] * kv[d][p];
        #pragma unroll
        for (int d2 = 0; d2 < 4; ++d2) g[d][d2] += qv[d][p] * kv[d2][p];
      }
  }

  float vals[24];
  #pragma unroll
  for (int c = 0; c < 4; ++c)
    #pragma unroll
    for (int d = 0; d < 4; ++d) vals[c * 4 + d] = g[c][d];
  #pragma unroll
  for (int d = 0; d < 4; ++d) { vals[16 + d] = qq[d]; vals[20 + d] = kk[d]; }

  __shared__ float sm[24];
  if (threadIdx.x < 24) sm[threadIdx.x] = 0.f;
  __syncthreads();
  #pragma unroll
  for (int j = 0; j < 24; ++j) {
    float v = vals[j];
    #pragma unroll
    for (int m = 16; m > 0; m >>= 1) v += __shfl_xor(v, m, 32);
    if ((threadIdx.x & 31) == 0) atomicAdd(&sm[j], v);
  }
  __syncthreads();
  if (threadIdx.x < 24) gram[bh * 24 + threadIdx.x] = sm[threadIdx.x];
}

// ---------------------------------------------------------------------------
// 5) attn[b,head,c,d] = softmax_d( G/(max(|q_c|,eps)*max(|k_d|,eps)) * temp )
// ---------------------------------------------------------------------------
__global__ void attn_kernel(const float* __restrict__ gram,
                            const float* __restrict__ temp,
                            float* __restrict__ attn) {
  const int t = threadIdx.x;                 // 128 threads: (b,head)
  const int b = t >> 4, head = t & 15;
  const float* g = gram + (b * NHEAD + head) * 24;
  const float tp = temp[head];
  float qn[4], kn[4];
  #pragma unroll
  for (int d = 0; d < 4; ++d) {
    qn[d] = fmaxf(sqrtf(g[16 + d]), 1e-12f);
    kn[d] = fmaxf(sqrtf(g[20 + d]), 1e-12f);
  }
  float* ap = attn + (b * NHEAD + head) * 16;
  #pragma unroll
  for (int c = 0; c < 4; ++c) {
    float a[4], mx = -3.4e38f;
    #pragma unroll
    for (int d = 0; d < 4; ++d) {
      a[d] = g[c * 4 + d] / (qn[c] * kn[d]) * tp;
      mx = fmaxf(mx, a[d]);
    }
    float sum = 0.f;
    #pragma unroll
    for (int d = 0; d < 4; ++d) { a[d] = __expf(a[d] - mx); sum += a[d]; }
    const float inv = 1.0f / sum;
    #pragma unroll
    for (int d = 0; d < 4; ++d) ap[c * 4 + d] = a[d] * inv;
  }
}

// ---------------------------------------------------------------------------
// 6) Fused attn@V + depthwise 3x3 ('SAME'):
//    t[c,n] = sum_d attn[head(c), c&3, d] * v[head(c)*4+d, n]; u = dw3x3(t).
//    Block = one (b, c, 8-row chunk). Halo tile of t built in LDS.
// ---------------------------------------------------------------------------
__global__ __launch_bounds__(256) void attnv_dw_kernel(const _Float16* __restrict__ qkv,
                                                       const float* __restrict__ attn,
                                                       const float* __restrict__ dww,
                                                       _Float16* __restrict__ u) {
  const int yb  = blockIdx.x & 15;
  const int bc  = blockIdx.x >> 4;
  const int b   = bc >> 6, cch = bc & 63;
  const int head = cch >> 2, row = cch & 3;
  const int y0  = yb << 3;

  __shared__ float T[10][WW];
  const float* ap = attn + ((b * NHEAD + head) * 4 + row) * 4;
  const float a0 = ap[0], a1 = ap[1], a2 = ap[2], a3 = ap[3];
  const _Float16* vp = qkv + ((size_t)b * NQKV + 2 * CC + head * 4) * HWSZ;

  for (int i = threadIdx.x; i < 10 * 64; i += 256) {   // pairs of columns
    const int ry = i >> 6, xc = (i & 63) << 1;
    const int gy = y0 + ry - 1;
    float t0 = 0.f, t1 = 0.f;
    if (gy >= 0 && gy < HH) {
      const int off = gy * WW + xc;
      #pragma unroll
      for (int d = 0; d < 4; ++d) {
        const v2h v2 = *(const v2h*)(vp + (size_t)d * HWSZ + off);
        const float ad = (d == 0) ? a0 : (d == 1) ? a1 : (d == 2) ? a2 : a3;
        t0 += ad * (float)v2[0];
        t1 += ad * (float)v2[1];
      }
    }
    T[ry][xc] = t0;
    T[ry][xc + 1] = t1;
  }
  __syncthreads();

  const float* wp = dww + cch * 9;
  float w9[9];
  #pragma unroll
  for (int j = 0; j < 9; ++j) w9[j] = wp[j];

  _Float16* up = u + ((size_t)b * CC + cch) * HWSZ;
  for (int i = threadIdx.x; i < 8 * 64; i += 256) {    // pairs of outputs
    const int oy = i >> 6, ox = (i & 63) << 1;
    v2h o2;
    #pragma unroll
    for (int p = 0; p < 2; ++p) {
      float acc = 0.f;
      #pragma unroll
      for (int ky = 0; ky < 3; ++ky)
        #pragma unroll
        for (int kx = 0; kx < 3; ++kx) {
          const int xx = ox + p + kx - 1;
          const float tv = (xx >= 0 && xx < WW) ? T[oy + ky][xx] : 0.f;
          acc += w9[ky * 3 + kx] * tv;
        }
      o2[p] = (_Float16)acc;
    }
    *(v2h*)(up + (y0 + oy) * WW + ox) = o2;
  }
}

// ---------------------------------------------------------------------------
// 7) Projection GEMM via WMMA, scaled by routing logit, accumulated into out:
//    out[b,o,n] (+)= logit[b,e] * sum_c pw[o,c] * u[b,c,n]
// ---------------------------------------------------------------------------
__global__ __launch_bounds__(256) void proj_gemm_kernel(const _Float16* __restrict__ u,
                                                        const float* __restrict__ pw,
                                                        const float* __restrict__ logit,
                                                        float* __restrict__ out,
                                                        int expert, int accumulate) {
  __shared__ __align__(16) _Float16 Us[128][XROW];  // pixel-major u tile
  __shared__ __align__(16) _Float16 Ps[16][CC];
  const int b    = blockIdx.x >> 7;
  const int pix0 = (blockIdx.x & 127) << 7;
  const int o0   = blockIdx.y << 4;
  const int tid  = threadIdx.x;

  for (int i = tid; i < 16 * CC; i += 256)
    Ps[i >> 6][i & 63] = (_Float16)pw[(o0 + (i >> 6)) * CC + (i & 63)];

  const _Float16* ub = u + (size_t)b * CC * HWSZ + pix0;
  for (int i = tid; i < 16 * 128; i += 256) {
    const int n = i & 127, c4 = (i >> 7) << 2;
    v4h v;
    #pragma unroll
    for (int j = 0; j < 4; ++j)
      v[j] = ub[(size_t)(c4 + j) * HWSZ + n];
    *(v4h*)&Us[n][c4] = v;
  }
  __syncthreads();

  const int lane = tid & 31;
  const int nloc = (tid >> 5) << 4;
  const int col  = lane & 15;
  const int hi   = (lane < 16) ? 0 : 8;
  const int kb   = (lane < 16) ? 0 : 16;

  v8f acc = {};
  #pragma unroll
  for (int ks = 0; ks < 2; ++ks) {
    const int c0 = ks * 32;
    const v16h a  = load_a_frag(&Ps[col][0], c0, kb);
    const v16h bf = load_b_frag(&Us[nloc + col][0], c0, kb);
    acc = __builtin_amdgcn_wmma_f32_16x16x32_f16(false, a, false, bf,
                                                 (short)0, acc, false, false);
  }

  const float lw = logit[b * EE + expert];
  float* op = out + (size_t)b * CC * HWSZ + (pix0 + nloc + col);
  #pragma unroll
  for (int r = 0; r < 8; ++r) {
    const size_t idx = (size_t)(o0 + r + hi) * HWSZ;
    float val = lw * acc[r];
    if (accumulate) val += op[idx];
    op[idx] = val;
  }
}

// ---------------------------------------------------------------------------
// Host-side launch
// ---------------------------------------------------------------------------
extern "C" void kernel_launch(void* const* d_in, const int* in_sizes, int n_in,
                              void* d_out, int out_size, void* d_ws, size_t ws_size,
                              hipStream_t stream) {
  (void)in_sizes; (void)n_in; (void)out_size; (void)ws_size;
  const float* x      = (const float*)d_in[0];
  const float* hidden = (const float*)d_in[1];
  const float* qkv_w  = (const float*)d_in[2];
  const float* dw_w   = (const float*)d_in[3];
  const float* proj_w = (const float*)d_in[4];
  const float* temp   = (const float*)d_in[5];
  const float* r1_w   = (const float*)d_in[6];
  const float* r1_b   = (const float*)d_in[7];
  const float* r3_w   = (const float*)d_in[8];
  const float* r3_b   = (const float*)d_in[9];

  float* out = (float*)d_out;
  char*  ws  = (char*)d_ws;
  float*    xg    = (float*)(ws + OFF_XG);
  float*    logit = (float*)(ws + OFF_LOGIT);
  float*    gram  = (float*)(ws + OFF_GRAM);
  float*    attn  = (float*)(ws + OFF_ATTN);
  _Float16* qkv   = (_Float16*)(ws + OFF_QKV);
  _Float16* u     = (_Float16*)(ws + OFF_U);

  mean_kernel<<<BB * CC, 256, 0, stream>>>(x, xg);
  routing_kernel<<<1, 128, 0, stream>>>(xg, hidden, r1_w, r1_b, r3_w, r3_b,
                                        out + OUT_HID_OFF, out + OUT_LOGIT_OFF, logit);

  for (int e = 0; e < EE; ++e) {
    qkv_gemm_kernel<<<dim3(BB * HWSZ / 128, NQKV / 16), 256, 0, stream>>>(
        x, qkv_w + (size_t)e * NQKV * CC, qkv);
    gram_kernel<<<BB * NHEAD, 256, 0, stream>>>(qkv, gram);
    attn_kernel<<<1, 128, 0, stream>>>(gram, temp + e * NHEAD, attn);
    attnv_dw_kernel<<<BB * CC * (HH / 8), 256, 0, stream>>>(
        qkv, attn, dw_w + (size_t)e * CC * 9, u);
    proj_gemm_kernel<<<dim3(BB * HWSZ / 128, CC / 16), 256, 0, stream>>>(
        u, proj_w + (size_t)e * CC * CC, logit, out, e, e > 0 ? 1 : 0);
  }
}
